// LayerPML_17935783428589
// MI455X (gfx1250) — matrix operand order repaired and checked by reading, not compiled, not run on hardware
//
#include <hip/hip_runtime.h>
#include <math.h>

// Problem constants (from reference): B=8, S_IN=4096, S_OUT=1024, D=128
#define BSZ   8
#define S_IN  4096
#define S_OUT 1024
#define DDIM  128
#define EPSV  1e-5f

typedef __attribute__((ext_vector_type(16))) _Float16 v16h;
typedef __attribute__((ext_vector_type(8)))  float    v8f;
typedef __attribute__((ext_vector_type(4)))  unsigned int u32x4;
typedef __attribute__((ext_vector_type(8)))  int      i32x8;
typedef __attribute__((ext_vector_type(4)))  int      i32x4;

union V16H { v16h v; _Float16 e[16]; };

#define WMMA_F16(A, Bm, C) \
  __builtin_amdgcn_wmma_f32_16x16x32_f16(false, (A), false, (Bm), (short)0, (C), false, false)

// ---------------------------------------------------------------------------
// Phase 1: xt = LayerNorm(x @ M^T) * gamma + beta
// One wave per 16 global rows (g = b*S_IN + s). WMMA 16x16x32 f16, K=128 in 4
// steps, 8 column tiles -> 8 f32 accumulators. C scattered to LDS (padded),
// then LayerNorm reduction per row and store f32 xt.
// ---------------------------------------------------------------------------
__global__ __launch_bounds__(32) void k_transform_ln(
    const float* __restrict__ x, const float* __restrict__ M,
    const float* __restrict__ gamma, const float* __restrict__ beta,
    float* __restrict__ xt)
{
  __shared__ float tile[16 * 132];   // pad 128->132 to kill bank conflicts
  const int lane = threadIdx.x;
  const int m    = lane & 15;
  const int half = lane >> 4;
  const long g0  = (long)blockIdx.x * 16;

  const float* xr = x + (g0 + m) * DDIM;

  v8f acc[8] = {};
  #pragma unroll
  for (int kb = 0; kb < 4; ++kb) {
    // A fragment: rows of x (f32 -> f16)
    V16H a;
    #pragma unroll
    for (int p = 0; p < 8; ++p) {
      a.e[p]     = (_Float16)xr[kb * 32 + 8 * half + p];
      a.e[8 + p] = (_Float16)xr[kb * 32 + 16 + 8 * half + p];
    }
    #pragma unroll
    for (int t = 0; t < 8; ++t) {
      // B fragment: B[k][n] = M[n][k] (M row-major, contiguous in k)
      const float* Mrow = M + (long)(t * 16 + m) * DDIM;
      V16H bm;
      #pragma unroll
      for (int j = 0; j < 8; ++j) {
        bm.e[j]     = (_Float16)Mrow[kb * 32 + 8 * half + j];
        bm.e[8 + j] = (_Float16)Mrow[kb * 32 + 16 + 8 * half + j];
      }
      acc[t] = WMMA_F16(a.v, bm.v, acc[t]);
    }
  }

  // Scatter C tiles to LDS: C[v + 8*half][t*16 + m]
  #pragma unroll
  for (int t = 0; t < 8; ++t)
    #pragma unroll
    for (int v = 0; v < 8; ++v)
      tile[(v + 8 * half) * 132 + t * 16 + m] = acc[t][v];
  __syncthreads();

  // LayerNorm: lanes 0..15 each own one row of 128
  if (lane < 16) {
    const float* row = &tile[lane * 132];
    float s = 0.f, s2 = 0.f;
    #pragma unroll 8
    for (int j = 0; j < DDIM; ++j) { float v = row[j]; s += v; s2 += v * v; }
    const float mu   = s * (1.f / DDIM);
    const float var  = s2 * (1.f / DDIM) - mu * mu;
    const float rstd = rsqrtf(var + EPSV);
    float* orow = xt + (g0 + lane) * DDIM;
    #pragma unroll 8
    for (int j = 0; j < DDIM; ++j)
      orow[j] = (row[j] - mu) * rstd * gamma[j] + beta[j];
  }
}

// ---------------------------------------------------------------------------
// Phase 1.5: LinkerT[t][s] = (f16) Linker[s][t]  (A-operand for phase 3)
// ---------------------------------------------------------------------------
__global__ __launch_bounds__(256) void k_linker_t(
    const float* __restrict__ Linker, _Float16* __restrict__ LT, int total)
{
  int idx = blockIdx.x * blockDim.x + threadIdx.x;
  if (idx >= total) return;
  int t = idx >> 12;       // / S_IN
  int s = idx & (S_IN - 1);
  LT[idx] = (_Float16)Linker[(long)s * S_OUT + t];
}

// ---------------------------------------------------------------------------
// Phase 2: NkT[b][i][s] = sum_j xt[b][s][j] * P[i][j] * cos(2*pi*s/(i*128+j+2))
// One block per s, thread = i. cos table amortized over all 8 batches via LDS.
// Output transposed (i-major) in f16 so phase-3 B fragments are contiguous.
// ---------------------------------------------------------------------------
__global__ __launch_bounds__(128) void k_modulate(
    const float* __restrict__ xt, const float* __restrict__ P,
    _Float16* __restrict__ NkT)
{
  __shared__ float xs[BSZ * DDIM];
  const int i = threadIdx.x;
  const int s = blockIdx.x;

  #pragma unroll
  for (int b = 0; b < BSZ; ++b)
    xs[b * DDIM + i] = xt[((long)b * S_IN + s) * DDIM + i];
  __syncthreads();

  const float* Pr = P + (long)i * DDIM;
  const float twopik = 6.283185307179586f * (float)s;

  float acc[BSZ] = {0.f};
  for (int j = 0; j < DDIM; ++j) {
    const float ph = cosf(twopik / (float)(i * DDIM + j + 2));
    const float c  = Pr[j] * ph;
    #pragma unroll
    for (int b = 0; b < BSZ; ++b) acc[b] += xs[b * DDIM + j] * c;
  }
  #pragma unroll
  for (int b = 0; b < BSZ; ++b)
    NkT[((long)b * DDIM + i) * S_IN + s] = (_Float16)acc[b];
}

// ---------------------------------------------------------------------------
// Phase 3: out[b] (1024x128) = Linker^T (1024x4096) @ Nk[b] (4096x128)
// One wave computes a 16(t) x 128(i) strip: 8 f32 accumulators, A fragment
// reused across all 8 WMMAs per k-step. A strip (16 x 128k, 4 KB f16) is
// staged into LDS by the Tensor Data Mover (tensor_load_to_lds), double-
// buffered and synchronized with s_wait_tensorcnt. B comes straight from the
// i-major NkT (two 16 B runs per lane, L2-resident).
// ---------------------------------------------------------------------------
#define KCHUNK 128                      // k elements per TDM tile
#define NCHUNK (S_IN / KCHUNK)          // 32 chunks

__device__ __forceinline__ void tdm_load_a(const _Float16* gsrc, _Float16* ldst)
{
  // D# group0: count=1 | lds_addr | global_addr[56:0] | type=2  (ISA 8.3)
  unsigned long long ga = (unsigned long long)gsrc;
  unsigned lds = (unsigned)(unsigned long long)ldst;   // low 32 bits = LDS offset
  u32x4 g0 = { 1u, lds, (unsigned)ga,
               ((unsigned)(ga >> 32) & 0x01FFFFFFu) | 0x80000000u };
  // D# group1 (ISA 8.4): data_size=1 (2B) @17:16; tensor_dim0=4096 @79:48;
  // tensor_dim1=1024 @111:80; tile_dim0=128 @127:112; tile_dim1=16 @143:128;
  // tensor_dim0_stride=4096 @207:160. All other fields 0.
  i32x8 g1 = { 0x00010000, 0x10000000, 0x04000000, 0x00800000,
               16, 4096, 0, 0 };
  i32x4 z4 = { 0, 0, 0, 0 };
#if __clang_major__ >= 23
  i32x8 z8 = { 0, 0, 0, 0, 0, 0, 0, 0 };
  __builtin_amdgcn_tensor_load_to_lds(g0, g1, z4, z4, z8, 0);
#else
  __builtin_amdgcn_tensor_load_to_lds(g0, g1, z4, z4, 0);
#endif
}

__global__ __launch_bounds__(32) void k_linkgemm(
    const _Float16* __restrict__ LT, const _Float16* __restrict__ NkT,
    float* __restrict__ out)
{
  __shared__ _Float16 abuf[2][16 * KCHUNK];   // 2 x 4 KB double buffer
  const int lane = threadIdx.x;
  const int m    = lane & 15;
  const int half = lane >> 4;
  const int tb   = blockIdx.x;  // t-tile 0..63
  const int b    = blockIdx.y;  // batch  0..7

  const _Float16* aTileBase = LT + (long)tb * 16 * S_IN;   // 16 rows, stride S_IN
  const _Float16* bBase     = NkT + ((long)b * DDIM + m) * S_IN;

  v8f acc[8] = {};

  tdm_load_a(aTileBase, &abuf[0][0]);                 // prime chunk 0

  for (int c = 0; c < NCHUNK; ++c) {
    const int buf = c & 1;
    if (c + 1 < NCHUNK) {
      tdm_load_a(aTileBase + (c + 1) * KCHUNK, &abuf[buf ^ 1][0]);
      __builtin_amdgcn_s_wait_tensorcnt(1);           // chunk c landed (in-order)
    } else {
      __builtin_amdgcn_s_wait_tensorcnt(0);
    }
    asm volatile("" ::: "memory");                    // LDS now holds chunk c

    const _Float16* arow = &abuf[buf][m * KCHUNK];
    #pragma unroll
    for (int ks = 0; ks < KCHUNK; ks += 32) {
      V16H a;
      #pragma unroll
      for (int p = 0; p < 8; ++p) {                   // ds_load_b128 x2
        a.e[p]     = arow[ks + 8 * half + p];
        a.e[8 + p] = arow[ks + 16 + 8 * half + p];
      }
      const long kg = (long)c * KCHUNK + ks;
      #pragma unroll
      for (int t = 0; t < 8; ++t) {
        const _Float16* brow = bBase + (long)t * 16 * S_IN + kg;
        V16H bm;
        #pragma unroll
        for (int j = 0; j < 8; ++j) {
          bm.e[j]     = brow[8 * half + j];
          bm.e[8 + j] = brow[16 + 8 * half + j];
        }
        acc[t] = WMMA_F16(a.v, bm.v, acc[t]);
      }
    }
  }

  float* obase = out + ((long)b * S_OUT + tb * 16) * DDIM;
  #pragma unroll
  for (int t = 0; t < 8; ++t)
    #pragma unroll
    for (int v = 0; v < 8; ++v)
      obase[(v + 8 * half) * DDIM + t * 16 + m] = acc[t][v];
}

// ---------------------------------------------------------------------------
extern "C" void kernel_launch(void* const* d_in, const int* in_sizes, int n_in,
                              void* d_out, int out_size, void* d_ws, size_t ws_size,
                              hipStream_t stream)
{
  (void)in_sizes; (void)n_in; (void)out_size; (void)ws_size;

  const float* x      = (const float*)d_in[0];
  const float* M      = (const float*)d_in[1];
  const float* P      = (const float*)d_in[2];
  const float* Linker = (const float*)d_in[3];
  const float* gamma  = (const float*)d_in[4];
  const float* beta   = (const float*)d_in[5];
  float* out = (float*)d_out;

  char* ws = (char*)d_ws;
  float*    xt  = (float*)ws;                                   // 16 MiB  (B*S_IN*D f32)
  _Float16* NkT = (_Float16*)(ws + (size_t)16 * 1024 * 1024);   //  8 MiB  (B*D*S_IN f16)
  _Float16* LT  = (_Float16*)(ws + (size_t)24 * 1024 * 1024);   //  8 MiB  (S_OUT*S_IN f16)

  // Phase 1: transform + LayerNorm (WMMA)
  k_transform_ln<<<(BSZ * S_IN) / 16, 32, 0, stream>>>(x, M, gamma, beta, xt);
  // Phase 1.5: transpose+convert Linker
  k_linker_t<<<(S_OUT * S_IN) / 256, 256, 0, stream>>>(Linker, LT, S_OUT * S_IN);
  // Phase 2: periodic modulation (cos amortized over batch)
  k_modulate<<<S_IN, DDIM, 0, stream>>>(xt, P, NkT);
  // Phase 3: Linker contraction (WMMA + TDM double-buffered A staging)
  k_linkgemm<<<dim3(S_OUT / 16, BSZ), 32, 0, stream>>>(LT, NkT, out);
}
